// GGNNModel_nextTimePred_33139967656252
// MI455X (gfx1250) — compile-verified, matrix operand order checked by reading, not run on hardware
//
#include <hip/hip_runtime.h>

// ---------------------------------------------------------------------------
// GGNN seq2seq decoder for MI455X (gfx1250).
// - Encoder is dead code in the reference (its state is never consumed): skip.
// - All matmuls: bf16 WMMA (v_wmma_f32_16x16x32_bf16), fp32 accumulate.
// - GEMM operands are kept in bf16 in global memory (weights/adjacency
//   converted once per launch; activations mirrored to bf16 by producers),
//   so tiles are staged with GLOBAL_LOAD_ASYNC_TO_LDS_B128 (ASYNCcnt) in a
//   double-buffered pipeline: no VGPR roundtrip, copy of tile k+1 overlaps
//   WMMA on tile k. Fragment gathers are explicit 128-bit LDS loads.
// ---------------------------------------------------------------------------

typedef __attribute__((ext_vector_type(16))) __bf16 v16bf;
typedef __attribute__((ext_vector_type(8)))  float  v8f;

#define GB  64          // batch
#define GN  512         // nodes
#define GU  128         // hidden
#define GT  12          // t_out
#define ROWS (GB * GN)  // 32768
#define SZ   (ROWS * GU)

#define BM 128
#define BN 128
#define BK 32
#define LDSROW 40       // BK + 8 pad (ushorts): 80B row stride, conflict-free gathers
#define STG_US 10240    // ushorts per stage (A 5120 + B 5120)
#define STG_B  20480    // bytes per stage

union BF16Frag { v16bf v; uint4 q[2]; unsigned int w[8]; unsigned short u[16]; };

__device__ __forceinline__ unsigned short f2bf(float f) {
  union { float f; unsigned int u; } x; x.f = f;
  unsigned int u = x.u;
  unsigned int r = u + 0x7FFFu + ((u >> 16) & 1u);   // round-to-nearest-even
  return (unsigned short)(r >> 16);
}

// C[M,Nn] (+)= bf16 A[M,K] @ bf16 B^T[Nn,K] (+bias[n]); batched via blockIdx.z.
// Optional bf16 outputs: O16 row-major mirror, OT per-batch-transposed
// (fixed geometry: batches of GN rows, GU cols -> OT[b][n][m]).
// M,K,Nn multiples of tile sizes (true for every call site).
__global__ __launch_bounds__(256)
void gemm_bf16_async(const unsigned short* __restrict__ A16,
                     const unsigned short* __restrict__ B16T,
                     float* __restrict__ C, const float* __restrict__ bias,
                     unsigned short* __restrict__ O16,
                     unsigned short* __restrict__ OT,
                     int M, int K, int Nn, int lda, int ldbT, int ldc,
                     long long sA, long long sB, long long sC, long long sO,
                     int beta)
{
  extern __shared__ unsigned short smem[];   // 2 stages x (A[128][40] + B[128][40])

  const int bz = blockIdx.z;
  A16 += (long long)bz * sA;
  B16T += (long long)bz * sB;
  if (C)   C   += (long long)bz * sC;
  if (O16) O16 += (long long)bz * sO;

  const int rowBlk = blockIdx.y * BM;
  const int colBlk = blockIdx.x * BN;

  const int tid  = threadIdx.x;
  const int lane = tid & 31;
  const int wave = tid >> 5;
  const int wr   = wave >> 1;    // 0..3 : 32-row group
  const int wc   = wave & 1;     // 0..1 : 64-col group
  const int half = lane >> 4;
  const int l16  = lane & 15;

  // issue one stage of async copies: exactly 4 instructions per wave
  // (2 for A tile, 2 for B tile; 16B chunk per lane per instruction).
  auto issue = [&](int stage, int kt) {
    const unsigned base = (unsigned)stage * STG_B;
#pragma unroll
    for (int it = 0; it < 2; ++it) {
      const int s   = tid + it * 256;          // chunk id 0..511
      const int row = s >> 2;                  // 0..127
      const int ch  = (s & 3) << 3;            // 0,8,16,24 (bf16 elems)
      const unsigned short* ga = A16 + (long long)(rowBlk + row) * lda + kt + ch;
      const unsigned la = base + (unsigned)(row * 80 + ch * 2);
      asm volatile("global_load_async_to_lds_b128 %0, %1, off"
                   :: "v"(la), "v"(ga) : "memory");
      const unsigned short* gb = B16T + (long long)(colBlk + row) * ldbT + kt + ch;
      const unsigned lb = base + 10240u + (unsigned)(row * 80 + ch * 2);
      asm volatile("global_load_async_to_lds_b128 %0, %1, off"
                   :: "v"(lb), "v"(gb) : "memory");
    }
  };

  v8f acc[2][4];
#pragma unroll
  for (int r = 0; r < 2; ++r)
#pragma unroll
    for (int c = 0; c < 4; ++c) acc[r][c] = {};

  const int nk = K / BK;
  issue(0, 0);

  for (int i = 0; i < nk; ++i) {
    const int cur = i & 1;
    if (i + 1 < nk) {
      issue(cur ^ 1, (i + 1) * BK);
      asm volatile("s_wait_asynccnt 4" ::: "memory");   // retire stage i only
    } else {
      asm volatile("s_wait_asynccnt 0" ::: "memory");
    }
    __syncthreads();

    const unsigned short* As = (const unsigned short*)
        __builtin_assume_aligned(smem + (size_t)cur * STG_US, 16);
    const unsigned short* Bs = As + 5120;    // +10240 bytes, still 16B aligned

    // fragment gathers (ISA 16-bit A-matrix 16x32 per-lane layout):
    // regs w[0..3] = 16 contiguous bytes at ushort offset half*8 of the row,
    // regs w[4..7] = 16 bytes at 16+half*8  -> two ds_load_b128 per fragment.
    BF16Frag af[2];
#pragma unroll
    for (int r = 0; r < 2; ++r) {
      const unsigned short* src = &As[(wr * 32 + r * 16 + l16) * LDSROW];
      af[r].q[0] = *(const uint4*)(src + half * 8);
      af[r].q[1] = *(const uint4*)(src + 16 + half * 8);
    }
    BF16Frag bfr[4];
#pragma unroll
    for (int c = 0; c < 4; ++c) {
      const unsigned short* src = &Bs[(wc * 64 + c * 16 + l16) * LDSROW];
      bfr[c].q[0] = *(const uint4*)(src + half * 8);
      bfr[c].q[1] = *(const uint4*)(src + 16 + half * 8);
    }

#pragma unroll
    for (int r = 0; r < 2; ++r)
#pragma unroll
      for (int c = 0; c < 4; ++c)
        acc[r][c] = __builtin_amdgcn_wmma_f32_16x16x32_bf16(
            false, af[r].v, false, bfr[c].v, (short)0, acc[r][c], false, false);

    __syncthreads();
  }

  // epilogue: D 16x16 f32 layout (vgpr v -> M = half*8+v, N = lane&15)
#pragma unroll
  for (int r = 0; r < 2; ++r) {
    const int mBase = rowBlk + wr * 32 + r * 16 + half * 8;
#pragma unroll
    for (int c = 0; c < 4; ++c) {
      const int n  = colBlk + wc * 64 + c * 16 + l16;
      const float bv = bias ? bias[n] : 0.0f;
#pragma unroll
      for (int v = 0; v < 8; ++v) {
        const int gm = mBase + v;
        const long long idx = (long long)gm * ldc + n;
        float val = acc[r][c][v] + bv;
        if (beta) val += C[idx];
        if (C)   C[idx]   = val;
        if (O16) O16[idx] = f2bf(val);
        if (OT) {                       // t^T[b][n][m] for the adjacency GEMM
          const int b = gm >> 9, ml = gm & (GN - 1);
          OT[(long long)b * (GN * GU) + (long long)n * GN + ml] = f2bf(val);
        }
      }
    }
  }
}

// --------------------------- elementwise kernels ---------------------------

__global__ void k_zero(float* x, int n) {
  int i = blockIdx.x * blockDim.x + threadIdx.x;
  if (i < n) x[i] = 0.0f;
}

__global__ void k_bsum(const float* __restrict__ be, float* __restrict__ bsum) {
  int u = threadIdx.x;
  bsum[u] = be[u] + be[GU + u];
}

// fp32 -> bf16 row-major copy
__global__ void k_cvt(const float* __restrict__ in, unsigned short* __restrict__ out, int n) {
  int i = blockIdx.x * blockDim.x + threadIdx.x;
  if (i < n) out[i] = f2bf(in[i]);
}

// fp32 [128,128] -> bf16 transposed [n][k]
__global__ void k_cvtT(const float* __restrict__ in, unsigned short* __restrict__ out) {
  int i = blockIdx.x * blockDim.x + threadIdx.x;
  if (i < GU * GU) {
    int k = i >> 7, n = i & (GU - 1);
    out[n * GU + k] = f2bf(in[i]);
  }
}

// h[row][u] = x[row]*Wp[u] + bp[u]  (+ bf16 mirror)
__global__ void k_proj_in1(const float* __restrict__ x, const float* __restrict__ Wp,
                           const float* __restrict__ bp, float* __restrict__ h,
                           unsigned short* __restrict__ h16) {
  int i = blockIdx.x * blockDim.x + threadIdx.x;
  if (i < SZ) {
    int u = i & (GU - 1);
    float v = x[i >> 7] * Wp[u] + bp[u];
    h[i] = v; h16[i] = f2bf(v);
  }
}

__device__ __forceinline__ float sigf(float x) { return 1.0f / (1.0f + expf(-x)); }

// z = sigmoid(z) in place; c16 = bf16(sigmoid(r) * h)
__global__ void k_sig_rh(float* __restrict__ z, const float* __restrict__ r,
                         const float* __restrict__ h, unsigned short* __restrict__ c16) {
  int i = blockIdx.x * blockDim.x + threadIdx.x;
  if (i < SZ) {
    z[i] = sigf(z[i]);
    c16[i] = f2bf(sigf(r[i]) * h[i]);
  }
}

// h = (1-z)*h + z*tanh(cpre)  (+ bf16 mirror)
__global__ void k_hupdate(float* __restrict__ h, const float* __restrict__ z,
                          const float* __restrict__ cpre, unsigned short* __restrict__ h16) {
  int i = blockIdx.x * blockDim.x + threadIdx.x;
  if (i < SZ) {
    float zz = z[i];
    float v = (1.0f - zz) * h[i] + zz * tanhf(cpre[i]);
    h[i] = v; h16[i] = f2bf(v);
  }
}

// out[b,t,n] = h[row]·W + b0 ; xnext[row] = same
__global__ void k_proj_out(const float* __restrict__ h, const float* __restrict__ W,
                           const float* __restrict__ b0, float* __restrict__ out,
                           float* __restrict__ xnext, int t) {
  int row = blockIdx.x * blockDim.x + threadIdx.x;
  if (row < ROWS) {
    const float* hr = h + (long long)row * GU;
    float s = b0[0];
#pragma unroll 8
    for (int u = 0; u < GU; ++u) s += hr[u] * W[u];
    int b = row >> 9, n = row & (GN - 1);
    out[(long long)b * (GT * GN) + (long long)t * GN + n] = s;
    xnext[row] = s;
  }
}

// ------------------------------ host helpers -------------------------------

struct Cell {
  const float *Wp, *bp, *We, *be, *Wz, *Uz, *bz, *Wr, *Ur, *br, *Wh, *Uh, *bh;
};
struct CellW {  // bf16, K-transposed weight mirrors
  const unsigned short *We0T, *We1T, *WzT, *UzT, *WrT, *UrT, *WhT, *UhT;
};

static void gemm(hipStream_t s, const unsigned short* A, const unsigned short* Bt,
                 float* C, const float* bias, unsigned short* O16, unsigned short* OT,
                 int M, int K, int Nn, int lda, int ldbT, int ldc,
                 long long sA, long long sB, long long sC, long long sO,
                 int batch, int beta) {
  dim3 g(Nn / BN, M / BM, batch), blk(256);
  gemm_bf16_async<<<g, blk, 2 * STG_B, s>>>(A, Bt, C, bias, O16, OT,
                                            M, K, Nn, lda, ldbT, ldc,
                                            sA, sB, sC, sO, beta);
}

// 3 GGNN propagation steps on h (fp32, in place) with bf16 mirror hb.
static void run_steps(hipStream_t s, const Cell& p, const CellW& w,
                      const unsigned short* adj16,
                      float* h, unsigned short* hb,
                      float* m, unsigned short* m16,
                      float* z, float* r, unsigned short* cb,
                      unsigned short* tT, float* bsum) {
  const dim3 eb(256), eg(SZ / 256);
  k_bsum<<<1, GU, 0, s>>>(p.be, bsum);
  const long long bs = (long long)GN * GU;   // per-batch stride (elements)
  for (int step = 0; step < 3; ++step) {
    // m = A0 @ (h @ We0) + bsum ; m += A1 @ (h @ We1)   (tT = bf16 (h@We)^T)
    gemm(s, hb, w.We0T, nullptr, nullptr, nullptr, tT,
         ROWS, GU, GU, GU, GU, GU, 0, 0, 0, 0, 1, 0);
    gemm(s, adj16, tT, m, bsum, nullptr, nullptr,
         GN, GN, GU, GN, GN, GU, 0, bs, bs, 0, GB, 0);
    gemm(s, hb, w.We1T, nullptr, nullptr, nullptr, tT,
         ROWS, GU, GU, GU, GU, GU, 0, 0, 0, 0, 1, 0);
    gemm(s, adj16 + (long long)GN * GN, tT, m, nullptr, m16, nullptr,
         GN, GN, GU, GN, GN, GU, 0, bs, bs, bs, GB, 1);
    // z = sig(m@Wz + h@Uz + bz) ; r = sig(m@Wr + h@Ur + br)
    gemm(s, m16, w.WzT, z, p.bz, nullptr, nullptr, ROWS, GU, GU, GU, GU, GU, 0,0,0,0, 1, 0);
    gemm(s, hb,  w.UzT, z, nullptr, nullptr, nullptr, ROWS, GU, GU, GU, GU, GU, 0,0,0,0, 1, 1);
    gemm(s, m16, w.WrT, r, p.br, nullptr, nullptr, ROWS, GU, GU, GU, GU, GU, 0,0,0,0, 1, 0);
    gemm(s, hb,  w.UrT, r, nullptr, nullptr, nullptr, ROWS, GU, GU, GU, GU, GU, 0,0,0,0, 1, 1);
    k_sig_rh<<<eg, eb, 0, s>>>(z, r, h, cb);
    // cpre (in r) = m@Wh + (sig(r)*h)@Uh + bh
    gemm(s, m16, w.WhT, r, p.bh, nullptr, nullptr, ROWS, GU, GU, GU, GU, GU, 0,0,0,0, 1, 0);
    gemm(s, cb,  w.UhT, r, nullptr, nullptr, nullptr, ROWS, GU, GU, GU, GU, GU, 0,0,0,0, 1, 1);
    k_hupdate<<<eg, eb, 0, s>>>(h, z, r, hb);
  }
}

// ---------------------------------------------------------------------------

extern "C" void kernel_launch(void* const* d_in, const int* in_sizes, int n_in,
                              void* d_out, int out_size, void* d_ws, size_t ws_size,
                              hipStream_t stream) {
  (void)in_sizes; (void)n_in; (void)out_size; (void)ws_size;
  // setup_inputs() insertion-order flattening:
  // 0 encoder_inputs, 1 decoder_inputs, 2 adjacency,
  // enc cell0 (13), enc cell1 (13), dec cell0 (13), dec cell1 (13), proj_W, proj_b.
  // Cell leaf order: Wp,bp,We,be,Wz,Uz,bz,Wr,Ur,br,Wh,Uh,bh
  const float* adj = (const float*)d_in[2];
  auto cell = [&](int base) {
    Cell c;
    c.Wp = (const float*)d_in[base + 0];  c.bp = (const float*)d_in[base + 1];
    c.We = (const float*)d_in[base + 2];  c.be = (const float*)d_in[base + 3];
    c.Wz = (const float*)d_in[base + 4];  c.Uz = (const float*)d_in[base + 5];
    c.bz = (const float*)d_in[base + 6];  c.Wr = (const float*)d_in[base + 7];
    c.Ur = (const float*)d_in[base + 8];  c.br = (const float*)d_in[base + 9];
    c.Wh = (const float*)d_in[base + 10]; c.Uh = (const float*)d_in[base + 11];
    c.bh = (const float*)d_in[base + 12];
    return c;
  };
  Cell dec0 = cell(29), dec1 = cell(42);   // encoder params (3..28) are dead
  const float* projW = (const float*)d_in[55];
  const float* projB = (const float*)d_in[56];
  float* out = (float*)d_out;

  // ---- workspace layout (~103 MB) ----
  char* p = (char*)d_ws;
  float* H  = (float*)p; p += (long long)SZ * 4;
  float* M_ = (float*)p; p += (long long)SZ * 4;
  float* Z  = (float*)p; p += (long long)SZ * 4;
  float* R  = (float*)p; p += (long long)SZ * 4;
  float* X  = (float*)p; p += (long long)ROWS * 4;
  float* bsum = (float*)p; p += 256 * 4;
  unsigned short* h16   = (unsigned short*)p; p += (long long)SZ * 2;
  unsigned short* m16   = (unsigned short*)p; p += (long long)SZ * 2;
  unsigned short* c16   = (unsigned short*)p; p += (long long)SZ * 2;
  unsigned short* tT    = (unsigned short*)p; p += (long long)SZ * 2;
  unsigned short* adj16 = (unsigned short*)p; p += (long long)2 * GN * GN * 2;
  unsigned short* wT    = (unsigned short*)p;

  // ---- one-time (per launch) bf16 conversions: adjacency + 17 weights ----
  k_cvt<<<(2 * GN * GN + 255) / 256, 256, 0, stream>>>(adj, adj16, 2 * GN * GN);
  const float* wsrc[17] = {
    dec0.We, dec0.We + GU * GU, dec0.Wz, dec0.Uz, dec0.Wr, dec0.Ur, dec0.Wh, dec0.Uh,
    dec1.We, dec1.We + GU * GU, dec1.Wz, dec1.Uz, dec1.Wr, dec1.Ur, dec1.Wh, dec1.Uh,
    dec1.Wp };
  for (int i = 0; i < 17; ++i)
    k_cvtT<<<(GU * GU + 255) / 256, 256, 0, stream>>>(wsrc[i], wT + (long long)i * GU * GU);
  auto wt = [&](int i) { return (const unsigned short*)(wT + (long long)i * GU * GU); };
  CellW w0 = { wt(0), wt(1), wt(2),  wt(3),  wt(4),  wt(5),  wt(6),  wt(7)  };
  CellW w1 = { wt(8), wt(9), wt(10), wt(11), wt(12), wt(13), wt(14), wt(15) };
  const unsigned short* Wp1T = wt(16);

  const dim3 eb(256), eg(SZ / 256);
  k_zero<<<(ROWS + 255) / 256, 256, 0, stream>>>(X, ROWS);   // go = zeros

  for (int t = 0; t < GT; ++t) {
    // ---- decoder cell 0 (in_dim = 1) : h in H, bf16 mirror h16 ----
    k_proj_in1<<<eg, eb, 0, stream>>>(X, dec0.Wp, dec0.bp, H, h16);
    run_steps(stream, dec0, w0, adj16, H, h16, M_, m16, Z, R, c16, tT, bsum);
    // ---- decoder cell 1: h1 = h0 @ Wp1 + bp1 (fp32 into Z, bf16 into c16) ----
    gemm(stream, h16, Wp1T, Z, dec1.bp, c16, nullptr,
         ROWS, GU, GU, GU, GU, GU, 0, 0, 0, 0, 1, 0);
    run_steps(stream, dec1, w1, adj16, Z, c16, M_, m16, /*z=*/H, R, /*cb=*/h16, tT, bsum);
    // ---- projection -> d_out[:, t, :] and next decoder input ----
    k_proj_out<<<(ROWS + 255) / 256, 256, 0, stream>>>(Z, projW, projB, out, X, t);
  }
}
